// Multihead_self_attention_21397527068764
// MI455X (gfx1250) — matrix operand order repaired and checked by reading, not compiled, not run on hardware
//
#include <hip/hip_runtime.h>
#include <hip/hip_bf16.h>

typedef __attribute__((ext_vector_type(16))) __bf16 v16bf;
typedef __attribute__((ext_vector_type(8)))  __bf16 v8bf;
typedef __attribute__((ext_vector_type(8)))  float  v8f;

#define BATCH  2
#define SEQ    2048
#define DMODEL 1024
#define NHEAD  16
#define HDIM   64
#define BSD    (BATCH * SEQ * DMODEL)   // 4,194,304 elements
#define DD     (DMODEL * DMODEL)        // 1,048,576 elements

static __device__ __forceinline__ __bf16 bf(float f) { return (__bf16)f; }

// combine two aligned 8x-bf16 halves into a 16x A/B fragment
static __device__ __forceinline__ v16bf ld16(const __bf16* p0, const __bf16* p1) {
  v8bf lo = *(const v8bf*)p0;
  v8bf hi = *(const v8bf*)p1;
  return __builtin_shufflevector(lo, hi, 0, 1, 2, 3, 4, 5, 6, 7,
                                 8, 9, 10, 11, 12, 13, 14, 15);
}

// reduce across the 16-lane half holding one C-fragment row
static __device__ __forceinline__ float halfmax(float v) {
  v = fmaxf(v, __shfl_xor(v, 1, 32));
  v = fmaxf(v, __shfl_xor(v, 2, 32));
  v = fmaxf(v, __shfl_xor(v, 4, 32));
  v = fmaxf(v, __shfl_xor(v, 8, 32));
  return v;
}
static __device__ __forceinline__ float halfsum(float v) {
  v += __shfl_xor(v, 1, 32);
  v += __shfl_xor(v, 2, 32);
  v += __shfl_xor(v, 4, 32);
  v += __shfl_xor(v, 8, 32);
  return v;
}

// shared 64x64-per-wave GEMM body: C[4][4] += A(64xK) * B(64xK)^T
#define GEMM_BODY(ARP, BRP, CACC)                                             \
  for (int k0 = 0; k0 < DMODEL; k0 += 32) {                                   \
    v16bf a0 = ld16(ARP[0] + k0 + sel * 16, ARP[0] + k0 + 8 + sel * 16);      \
    v16bf a1 = ld16(ARP[1] + k0 + sel * 16, ARP[1] + k0 + 8 + sel * 16);      \
    v16bf a2 = ld16(ARP[2] + k0 + sel * 16, ARP[2] + k0 + 8 + sel * 16);      \
    v16bf a3 = ld16(ARP[3] + k0 + sel * 16, ARP[3] + k0 + 8 + sel * 16);      \
    const int bo = k0 + sel * 16;                                             \
    _Pragma("unroll") for (int j = 0; j < 4; ++j) {                           \
      v16bf bb = *(const v16bf*)(BRP[j] + bo);                                \
      CACC[0][j] = __builtin_amdgcn_wmma_f32_16x16x32_bf16(                   \
          false, a0, false, bb, (short)0, CACC[0][j], false, false);          \
      CACC[1][j] = __builtin_amdgcn_wmma_f32_16x16x32_bf16(                   \
          false, a1, false, bb, (short)0, CACC[1][j], false, false);          \
      CACC[2][j] = __builtin_amdgcn_wmma_f32_16x16x32_bf16(                   \
          false, a2, false, bb, (short)0, CACC[2][j], false, false);          \
      CACC[3][j] = __builtin_amdgcn_wmma_f32_16x16x32_bf16(                   \
          false, a3, false, bb, (short)0, CACC[3][j], false, false);          \
    }                                                                         \
  }

// ---------------------------------------------------------------------------
// Kernel 0: one-shot fp32 -> bf16 conversion (8 elements / thread)
// ---------------------------------------------------------------------------
__global__ __launch_bounds__(256)
void cvt_bf16_kernel(const float* __restrict__ src, __bf16* __restrict__ dst,
                     int n8) {
  int i = blockIdx.x * 256 + threadIdx.x;
  if (i >= n8) return;
  const float4* s = (const float4*)src + (size_t)i * 2;
  float4 f0 = s[0], f1 = s[1];
  v8bf o;
  o[0] = bf(f0.x); o[1] = bf(f0.y); o[2] = bf(f0.z); o[3] = bf(f0.w);
  o[4] = bf(f1.x); o[5] = bf(f1.y); o[6] = bf(f1.z); o[7] = bf(f1.w);
  *(v8bf*)(dst + (size_t)i * 8) = o;
}

// ---------------------------------------------------------------------------
// Kernel 1a: q/k = x @ W^T with fused RoPE and 1/sqrt(hd) scale on q.
// One wave computes a 64x64 block = 4x4 WMMA tiles; N block == one head.
// Output bf16 [B,H,S,hd].
// ---------------------------------------------------------------------------
__global__ __launch_bounds__(32)
void qk_gemm_rope_kernel(const __bf16* __restrict__ xb,
                         const __bf16* __restrict__ wqb,
                         const __bf16* __restrict__ wkb,
                         __bf16* __restrict__ qo,
                         __bf16* __restrict__ ko) {
  const int lane  = threadIdx.x;
  const int mt    = blockIdx.x;   // 64-row block over B*S
  const int h     = blockIdx.y;   // head == 64-col block
  const int which = blockIdx.z;   // 0=q 1=k
  const __bf16* __restrict__ W = (which == 0) ? wqb : wkb;
  __bf16* __restrict__ outp    = (which == 0) ? qo  : ko;

  const int mrow = lane & 15;
  const int sel  = lane >> 4;

  const __bf16* ar[4];
  const __bf16* br[4];
#pragma unroll
  for (int i = 0; i < 4; ++i) {
    ar[i] = xb + (size_t)(mt * 64 + i * 16 + mrow) * DMODEL;
    br[i] = W  + (size_t)(h  * 64 + i * 16 + mrow) * DMODEL;
  }

  v8f c[4][4] = {};
  GEMM_BODY(ar, br, c)

  const float qscale = (which == 0) ? 0.125f : 1.0f;  // 1/sqrt(64)
#pragma unroll
  for (int j = 0; j < 4; ++j) {
    const int  hc     = j * 16 + mrow;   // column within head
    const bool iseven = (hc & 1) == 0;
    const float invf  = __expf(-(float)(hc & ~1) * (9.210340372f / 64.0f));
#pragma unroll
    for (int i = 0; i < 4; ++i) {
      const int gmb = mt * 64 + i * 16;
      const int bb  = gmb >> 11;                      // batch (tile-uniform)
      const int ssb = (gmb & (SEQ - 1)) + 8 * sel;    // base seq pos for lane
      __bf16* p = outp + (((size_t)bb * NHEAD + h) * SEQ + ssb) * HDIM + hc;
      const float posb = (float)ssb;
#pragma unroll
      for (int r = 0; r < 8; ++r) {
        float val = c[i][j][r];
        float sn, cs;
        __sincosf((posb + (float)r) * invf, &sn, &cs);
        float other = __shfl_xor(val, 1, 32);  // even<->odd partner column
        val = iseven ? (val * cs - other * sn) : (other * sn + val * cs);
        p[r * HDIM] = bf(val * qscale);        // constant-offset stores
      }
    }
  }
}

// ---------------------------------------------------------------------------
// Kernel 1b: v = x @ Wv^T, written TRANSPOSED as bf16 [B,H,hd,S].
// Per (i,j) fragment one lane holds 8 consecutive S positions -> single
// 16B store per fragment.
// ---------------------------------------------------------------------------
__global__ __launch_bounds__(32)
void v_gemm_kernel(const __bf16* __restrict__ xb,
                   const __bf16* __restrict__ wvb,
                   __bf16* __restrict__ vo) {
  const int lane = threadIdx.x;
  const int mt   = blockIdx.x;
  const int h    = blockIdx.y;
  const int mrow = lane & 15;
  const int sel  = lane >> 4;

  const __bf16* ar[4];
  const __bf16* br[4];
#pragma unroll
  for (int i = 0; i < 4; ++i) {
    ar[i] = xb  + (size_t)(mt * 64 + i * 16 + mrow) * DMODEL;
    br[i] = wvb + (size_t)(h  * 64 + i * 16 + mrow) * DMODEL;
  }

  v8f c[4][4] = {};
  GEMM_BODY(ar, br, c)

#pragma unroll
  for (int j = 0; j < 4; ++j) {
    const int hc = j * 16 + mrow;
#pragma unroll
    for (int i = 0; i < 4; ++i) {
      const int gmb = mt * 64 + i * 16;
      const int bb  = gmb >> 11;
      const int ssb = (gmb & (SEQ - 1)) + 8 * sel;
      __bf16* p = vo + (((size_t)bb * NHEAD + h) * HDIM + hc) * SEQ + ssb;
      v8bf pk;
#pragma unroll
      for (int r = 0; r < 8; ++r) pk[r] = bf(c[i][j][r]);
      *(v8bf*)p = pk;   // one b128 store per fragment
    }
  }
}

// ---------------------------------------------------------------------------
// Kernel 2: causal flash attention, one wave per (b,h, 16-query tile).
// q,k in [B,H,S,hd] bf16; v in [B,H,hd,S] bf16 (pre-transposed) -> all
// B-fragments are contiguous 32B loads. LDS only for the P transpose.
// ---------------------------------------------------------------------------
__global__ __launch_bounds__(32)
void flash_attn_kernel(const __bf16* __restrict__ q,
                       const __bf16* __restrict__ k,
                       const __bf16* __restrict__ v,
                       __bf16* __restrict__ y) {
  __shared__ __bf16 Pl[16][40];  // P tile (16 q x 32 keys), 80B rows

  const int lane = threadIdx.x;
  const int qi   = blockIdx.x;   // query tile 0..127
  const int bh   = blockIdx.y;   // 0..31
  const int mrow = lane & 15;
  const int sel  = lane >> 4;

  const __bf16* __restrict__ Q  = q + (size_t)bh * SEQ * HDIM;
  const __bf16* __restrict__ K  = k + (size_t)bh * SEQ * HDIM;
  const __bf16* __restrict__ Vt = v + (size_t)bh * HDIM * SEQ;  // [hd][S]

  // Q A-fragments (pre-scaled by 1/sqrt(hd) in the projection)
  v16bf aq[2];
  {
    const __bf16* qr = Q + (size_t)(qi * 16 + mrow) * HDIM;
    aq[0] = ld16(qr + sel * 16,      qr + 8 + sel * 16);
    aq[1] = ld16(qr + 32 + sel * 16, qr + 40 + sel * 16);
  }

  // loop-invariant fragment base pointers
  const __bf16* kp0 = K + (size_t)mrow * HDIM + sel * 16;         // + kbase*HDIM
  const __bf16* kp1 = K + (size_t)(16 + mrow) * HDIM + sel * 16;
  const __bf16* vp[4];
#pragma unroll
  for (int t = 0; t < 4; ++t)
    vp[t] = Vt + (size_t)(t * 16 + mrow) * SEQ + sel * 16;        // + kbase

  float mi[8], li[8];
  v8f   o[4] = {};
#pragma unroll
  for (int r = 0; r < 8; ++r) { mi[r] = -1e30f; li[r] = 0.0f; }

  const int nkb = (qi >> 1) + 1;  // causal: key blocks covering keys <= qmax
  for (int kb = 0; kb < nkb; ++kb) {
    const int kbase = kb * 32;

    // scores: s0 = keys kbase..+15, s1 = keys kbase+16..+31
    v8f s0 = {}, s1 = {};
#pragma unroll
    for (int c = 0; c < 2; ++c) {
      v16bf bk0 = *(const v16bf*)(kp0 + (size_t)kbase * HDIM + c * 32);
      v16bf bk1 = *(const v16bf*)(kp1 + (size_t)kbase * HDIM + c * 32);
      s0 = __builtin_amdgcn_wmma_f32_16x16x32_bf16(false, aq[c], false, bk0,
                                                   (short)0, s0, false, false);
      s1 = __builtin_amdgcn_wmma_f32_16x16x32_bf16(false, aq[c], false, bk1,
                                                   (short)0, s1, false, false);
    }

    // causal mask + online softmax (rows live in 16-lane halves)
    const int qg0 = qi * 16;
#pragma unroll
    for (int r = 0; r < 8; ++r) {
      const int qrg = qg0 + r + 8 * sel;
      const int kc0 = kbase + mrow;
      const int kc1 = kbase + 16 + mrow;
      float s0r = (kc0 <= qrg) ? s0[r] : -1e30f;
      float s1r = (kc1 <= qrg) ? s1[r] : -1e30f;
      float rm  = halfmax(fmaxf(s0r, s1r));
      float mn  = fmaxf(mi[r], rm);
      float p0  = __expf(s0r - mn);
      float p1  = __expf(s1r - mn);
      float al  = __expf(mi[r] - mn);
      float rs  = halfsum(p0 + p1);
      li[r] = li[r] * al + rs;
      mi[r] = mn;
      o[0][r] *= al; o[1][r] *= al; o[2][r] *= al; o[3][r] *= al;
      Pl[r + 8 * sel][mrow]      = bf(p0);   // C-layout -> LDS (row, col)
      Pl[r + 8 * sel][16 + mrow] = bf(p1);
    }
    __syncthreads();

    // P A-fragment from LDS (vectorized 16B reads)
    v16bf ap = ld16(&Pl[mrow][sel * 16], &Pl[mrow][8 + sel * 16]);

    // PV: B fragment = 16 consecutive keys of a V^T row
#pragma unroll
    for (int t = 0; t < 4; ++t) {
      v16bf bv = *(const v16bf*)(vp[t] + kbase);
      o[t] = __builtin_amdgcn_wmma_f32_16x16x32_bf16(false, ap, false, bv,
                                                     (short)0, o[t], false, false);
    }
    __syncthreads();
  }

  // normalize and write y (bf16, [B,S,D] layout)
  const int b = bh >> 4, h = bh & 15;
  __bf16* yb = y + ((size_t)b * SEQ + qi * 16 + 8 * sel) * DMODEL + h * HDIM + mrow;
#pragma unroll
  for (int r = 0; r < 8; ++r) {
    float inv = 1.0f / li[r];
    yb[r * DMODEL + 0]  = bf(o[0][r] * inv);
    yb[r * DMODEL + 16] = bf(o[1][r] * inv);
    yb[r * DMODEL + 32] = bf(o[2][r] * inv);
    yb[r * DMODEL + 48] = bf(o[3][r] * inv);
  }
}

// ---------------------------------------------------------------------------
// Kernel 3: out = y @ Wo^T, 64x64 per wave, fp32 output.
// ---------------------------------------------------------------------------
__global__ __launch_bounds__(32)
void out_proj_kernel(const __bf16* __restrict__ y,
                     const __bf16* __restrict__ wob,
                     float* __restrict__ out) {
  const int lane = threadIdx.x;
  const int mt   = blockIdx.x;
  const int nt   = blockIdx.y;
  const int mrow = lane & 15;
  const int sel  = lane >> 4;

  const __bf16* ar[4];
  const __bf16* br[4];
#pragma unroll
  for (int i = 0; i < 4; ++i) {
    ar[i] = y   + (size_t)(mt * 64 + i * 16 + mrow) * DMODEL;
    br[i] = wob + (size_t)(nt * 64 + i * 16 + mrow) * DMODEL;
  }

  v8f c[4][4] = {};
  GEMM_BODY(ar, br, c)

#pragma unroll
  for (int i = 0; i < 4; ++i) {
    float* p = out + (size_t)(mt * 64 + i * 16 + 8 * sel) * DMODEL +
               nt * 64 + mrow;
#pragma unroll
    for (int r = 0; r < 8; ++r) {
#pragma unroll
      for (int j = 0; j < 4; ++j) p[r * DMODEL + j * 16] = c[i][j][r];
    }
  }
}

// ---------------------------------------------------------------------------
extern "C" void kernel_launch(void* const* d_in, const int* in_sizes, int n_in,
                              void* d_out, int out_size, void* d_ws, size_t ws_size,
                              hipStream_t stream) {
  (void)in_sizes; (void)n_in; (void)out_size; (void)ws_size;
  const float* x  = (const float*)d_in[0];
  const float* Wq = (const float*)d_in[1];
  const float* Wk = (const float*)d_in[2];
  const float* Wv = (const float*)d_in[3];
  const float* Wo = (const float*)d_in[4];

  __bf16* ws  = (__bf16*)d_ws;
  __bf16* xb  = ws;                              // [B*S, D]
  __bf16* wqb = ws + (size_t)BSD;                // [D, D]
  __bf16* wkb = wqb + (size_t)DD;
  __bf16* wvb = wkb + (size_t)DD;
  __bf16* wob = wvb + (size_t)DD;
  __bf16* qb  = wob + (size_t)DD;                // [B,H,S,hd]
  __bf16* kb  = qb + (size_t)BSD;
  __bf16* vb  = kb + (size_t)BSD;                // [B,H,hd,S] (transposed)
  __bf16* yb  = vb + (size_t)BSD;                // [B,S,D]

  // prep: fp32 -> bf16 staging
  cvt_bf16_kernel<<<BSD / 8 / 256, 256, 0, stream>>>(x,  xb,  BSD / 8);
  cvt_bf16_kernel<<<DD  / 8 / 256, 256, 0, stream>>>(Wq, wqb, DD / 8);
  cvt_bf16_kernel<<<DD  / 8 / 256, 256, 0, stream>>>(Wk, wkb, DD / 8);
  cvt_bf16_kernel<<<DD  / 8 / 256, 256, 0, stream>>>(Wv, wvb, DD / 8);
  cvt_bf16_kernel<<<DD  / 8 / 256, 256, 0, stream>>>(Wo, wob, DD / 8);

  dim3 gqk(BATCH * SEQ / 64, NHEAD, 2);
  qk_gemm_rope_kernel<<<gqk, 32, 0, stream>>>(xb, wqb, wkb, qb, kb);

  dim3 gv(BATCH * SEQ / 64, NHEAD);
  v_gemm_kernel<<<gv, 32, 0, stream>>>(xb, wvb, vb);

  dim3 gattn(SEQ / 16, BATCH * NHEAD);
  flash_attn_kernel<<<gattn, 32, 0, stream>>>(qb, kb, vb, yb);

  dim3 gout(BATCH * SEQ / 64, DMODEL / 64);
  out_proj_kernel<<<gout, 32, 0, stream>>>(yb, wob, (float*)d_out);
}
